// SwinTransformerBlock_12421045420295
// MI455X (gfx1250) — matrix operand order repaired and checked by reading, not compiled, not run on hardware
//
#include <hip/hip_runtime.h>
#include <hip/hip_bf16.h>
#include <math.h>

// ---- problem constants (Swin block) ----
#define BATCH   32
#define HH      56
#define WW      56
#define CC      192
#define WS      7
#define SHIFT   3
#define NH      6
#define HD      32          // C / NH
#define NTOK    49          // WS*WS
#define NPAD    64          // padded tokens per window
#define NWIN    64          // (H/WS)*(W/WS)
#define BNW     (BATCH*NWIN)        // 2048 windows
#define MROWS   (BNW*NPAD)          // 131072 padded window rows
#define XROWS   (BATCH*HH*WW)       // 100352 image rows
#define SCALE   0.17677669529663687f  // HD^-0.5

typedef __attribute__((ext_vector_type(16))) _Float16 v16h;
typedef __attribute__((ext_vector_type(8)))  _Float16 v8h;
typedef __attribute__((ext_vector_type(8)))  float    v8f;
typedef __attribute__((ext_vector_type(4)))  unsigned int v4u;
typedef __attribute__((ext_vector_type(8)))  int      v8i;
typedef __attribute__((ext_vector_type(4)))  int      v4i;

static __device__ __forceinline__ v16h cat16(v8h lo, v8h hi) {
    return __builtin_shufflevector(lo, hi, 0,1,2,3,4,5,6,7,8,9,10,11,12,13,14,15);
}

// ---------------------------------------------------------------------------
// Tensor Data Mover: 2D tile (tile_w elems x tile_h rows) global -> LDS.
// D# per CDNA5 ISA:
//  group0: [1:0]=count=1, [63:32]=lds_addr, [120:64]=global_addr, [127:126]=2
//  group1: [17:16]=data_size(1 => 2B), [79:48]=tensor_dim0, [111:80]=tensor_dim1,
//          [127:112]=tile_dim0, [143:128]=tile_dim1, [207:160]=dim0_stride
// Issue from ONE wave only (EXEC ignored; op is per-wave). Track w/ TENSORcnt.
// ---------------------------------------------------------------------------
static __device__ __forceinline__ void tdm_load_2d_f16(unsigned int lds_off,
                                                       const _Float16* gptr,
                                                       unsigned int tile_w,
                                                       unsigned int tile_h,
                                                       unsigned int stride_elems) {
    unsigned long long ga = (unsigned long long)(uintptr_t)gptr;
    v4u g0;
    g0.x = 1u;                                          // count=1, load, user desc
    g0.y = lds_off;                                     // lds_addr
    g0.z = (unsigned int)ga;                            // global_addr[31:0]
    g0.w = (unsigned int)((ga >> 32) & 0x01FFFFFFu) | (2u << 30);  // addr[56:32] | type=2
    v8i g1;
    g1[0] = (int)0x10000u;                              // data_size=1 (2 bytes)
    g1[1] = (int)(tile_w << 16);                        // tensor_dim0[15:0] in [31:16]
    g1[2] = (int)(tile_h << 16);                        // tensor_dim1[15:0] in [31:16]
    g1[3] = (int)(tile_w << 16);                        // tile_dim0 in [31:16]
    g1[4] = (int)tile_h;                                // tile_dim1 (tile_dim2=0)
    g1[5] = (int)stride_elems;                          // tensor_dim0_stride[31:0]
    g1[6] = 0;
    g1[7] = 0;
    v4i z = {0, 0, 0, 0};
#if defined(__clang_major__) && (__clang_major__ >= 23)
    v8i z8 = {0, 0, 0, 0, 0, 0, 0, 0};
    __builtin_amdgcn_tensor_load_to_lds(g0, g1, z, z, z8, 0);
#else
    __builtin_amdgcn_tensor_load_to_lds(g0, g1, z, z, 0);
#endif
}

// ---------------------------------------------------------------------------
// f32 -> f16 weight convert
// ---------------------------------------------------------------------------
__global__ void k_f32_to_f16(const float* __restrict__ in, _Float16* __restrict__ out, int n) {
    int i = blockIdx.x * 256 + threadIdx.x;
    if (i < n) out[i] = (_Float16)in[i];
}

// ---------------------------------------------------------------------------
// LN1 + cyclic shift + window partition (49 -> 64 padded rows), f16 out
// grid: BNW*NPAD blocks, 192 threads (one per channel)
// ---------------------------------------------------------------------------
__global__ void k_ln1_partition(const float* __restrict__ x,
                                const float* __restrict__ g,
                                const float* __restrict__ b,
                                _Float16* __restrict__ hwin) {
    int bid = blockIdx.x;
    int w   = bid >> 6;          // window id 0..2047
    int n   = bid & 63;          // padded token id
    int c   = threadIdx.x;       // channel
    size_t outIdx = ((size_t)w * NPAD + n) * CC + c;
    if (n >= NTOK) { hwin[outIdx] = (_Float16)0.f; return; }

    int bimg = w >> 6;
    int wi   = (w >> 3) & 7;
    int wj   = w & 7;
    int r = n / WS, cc = n % WS;
    int sh = wi * WS + r, sw = wj * WS + cc;             // shifted-image coords
    int oh = (sh + SHIFT) % HH, ow = (sw + SHIFT) % WW;  // original coords
    float v = x[((size_t)bimg * (HH * WW) + oh * WW + ow) * CC + c];

    float s = v, s2 = v * v;
    #pragma unroll
    for (int o = 16; o >= 1; o >>= 1) { s += __shfl_xor(s, o, 32); s2 += __shfl_xor(s2, o, 32); }
    __shared__ float rs[6], rq[6];
    int wv = c >> 5;
    if ((c & 31) == 0) { rs[wv] = s; rq[wv] = s2; }
    __syncthreads();
    float ts = 0.f, tq = 0.f;
    #pragma unroll
    for (int i = 0; i < 6; i++) { ts += rs[i]; tq += rq[i]; }
    float mu  = ts * (1.f / CC);
    float var = tq * (1.f / CC) - mu * mu;
    float y = (v - mu) * rsqrtf(var + 1e-5f) * g[c] + b[c];
    hwin[outIdx] = (_Float16)y;
}

// ---------------------------------------------------------------------------
// WMMA GEMM with TDM double-buffered A staging:
//   C[M,N] = A[M,K] @ Bw[N,K]^T + bias, epilogues:
// MODE 0: f16 out, bias        (qkv)
// MODE 1: f16 out, bias+GELU   (fc1)
// MODE 2: f32 out, bias+resid  (fc2 -> d_out)
// MODE 3: f32 out, bias        (proj)
// block: 256 thr = 8 waves; wave -> 16x64 strip; block tile 128x64.
// M % 128 == 0, N % 64 == 0, K % 32 == 0 (true for all uses here).
// Wave 0 drives the TDM; s_wait_tensorcnt + barrier publish each K-slab.
// ---------------------------------------------------------------------------
template <int MODE>
__global__ void k_wmma_gemm(const _Float16* __restrict__ A,
                            const _Float16* __restrict__ Bw,
                            const float*    __restrict__ bias,
                            void*           __restrict__ Cout,
                            const float*    __restrict__ resid,
                            int M, int N, int K) {
    __shared__ __align__(32) _Float16 Abuf[2][128][32];   // 16 KB double buffer

    int lane = threadIdx.x & 31;
    int wv   = threadIdx.x >> 5;               // 0..7
    int half = lane & 15;
    int hi   = lane >> 4;                      // 0 or 1
    int rowTile = blockIdx.y * 128 + wv * 16;
    int colBase = blockIdx.x * 64;

    const _Float16* Ablk = A + (size_t)(blockIdx.y * 128) * K;
    int nIt = K >> 5;

    if (wv == 0) {
        tdm_load_2d_f16((unsigned int)(uintptr_t)&Abuf[0][0][0], Ablk, 32, 128, (unsigned int)K);
    }

    v8f acc[4] = {};
    for (int it = 0; it < nIt; ++it) {
        int k0 = it << 5;
        if (wv == 0) __builtin_amdgcn_s_wait_tensorcnt((short)0);
        __syncthreads();                       // slab `it` resident; prior readers done
        if (wv == 0 && it + 1 < nIt) {
            tdm_load_2d_f16((unsigned int)(uintptr_t)&Abuf[(it + 1) & 1][0][0],
                            Ablk + (size_t)(k0 + 32), 32, 128, (unsigned int)K);
        }
        const _Float16* arow = &Abuf[it & 1][wv * 16 + half][hi ? 8 : 0];
        v16h a = cat16(*(const v8h*)arow, *(const v8h*)(arow + 16));
        #pragma unroll
        for (int t = 0; t < 4; t++) {
            const _Float16* brow = Bw + (size_t)(colBase + 16 * t + half) * K + k0 + (hi ? 8 : 0);
            v16h bfrag = cat16(*(const v8h*)brow, *(const v8h*)(brow + 16));
            acc[t] = __builtin_amdgcn_wmma_f32_16x16x32_f16(
                false, a, false, bfrag, (short)0, acc[t], false, false);
        }
    }

    int rbase = rowTile + hi * 8;
    #pragma unroll
    for (int t = 0; t < 4; t++) {
        int cn = colBase + 16 * t + half;
        float bv = bias[cn];
        #pragma unroll
        for (int j = 0; j < 8; j++) {
            size_t idx = (size_t)(rbase + j) * N + cn;
            float v = acc[t][j] + bv;
            if (MODE == 1) v = 0.5f * v * (1.f + erff(v * 0.70710678118654752f));
            if (MODE == 0 || MODE == 1) {
                ((_Float16*)Cout)[idx] = (_Float16)v;
            } else if (MODE == 2) {
                ((float*)Cout)[idx] = v + resid[idx];
            } else {
                ((float*)Cout)[idx] = v;
            }
        }
    }
}

// ---------------------------------------------------------------------------
// Windowed attention, one block per (window, head); 128 thr = 4 waves.
// qkv layout per padded row (576): [q(6*32) | k(6*32) | v(6*32)]
// ---------------------------------------------------------------------------
__global__ void k_window_attention(const _Float16* __restrict__ qkv,
                                   const float* __restrict__ rpb,
                                   _Float16* __restrict__ h2) {
    int w = blockIdx.x;                 // window 0..2047
    int h = blockIdx.y;                 // head 0..5
    int tid = threadIdx.x;
    int lane = tid & 31, wv = tid >> 5;
    int half = lane & 15, hi = lane >> 4;

    __shared__ __align__(32) _Float16 VT[HD][80];      // V transposed [dim][token]
    __shared__ __align__(32) _Float16 P[NPAD][80];     // softmax probs
    __shared__ float RPB[(2 * WS - 1) * (2 * WS - 1)]; // head-h rel-pos bias (169)

    const _Float16* base = qkv + (size_t)w * NPAD * (3 * CC);
    const _Float16* Qg = base + h * HD;
    const _Float16* Kg = base + CC + h * HD;
    const _Float16* Vg = base + 2 * CC + h * HD;

    // stage V^T and head-specific bias table into LDS
    for (int i = tid; i < NPAD * HD; i += 128) {
        int t = i >> 5, d = i & 31;
        VT[d][t] = Vg[(size_t)t * (3 * CC) + d];
    }
    for (int i = tid; i < (2 * WS - 1) * (2 * WS - 1); i += 128) {
        RPB[i] = rpb[i * NH + h];
    }
    __syncthreads();

    // S = Q @ K^T  (wave wv owns query rows 16wv..16wv+15)
    v8f S[4] = {};
    {
        const _Float16* arow = Qg + (size_t)(16 * wv + half) * (3 * CC) + (hi ? 8 : 0);
        v16h a = cat16(*(const v8h*)arow, *(const v8h*)(arow + 16));
        #pragma unroll
        for (int t = 0; t < 4; t++) {
            const _Float16* brow = Kg + (size_t)(16 * t + half) * (3 * CC) + (hi ? 8 : 0);
            v16h bfrag = cat16(*(const v8h*)brow, *(const v8h*)(brow + 16));
            S[t] = __builtin_amdgcn_wmma_f32_16x16x32_f16(
                false, a, false, bfrag, (short)0, S[t], false, false);
        }
    }

    // scale + rel-pos bias + shift mask + row softmax (per-half shfl reduce)
    int wi = (w >> 3) & 7, wj = w & 7;
    int m_col = 16 * 0 + half;   // key token column base per tile computed in loop
    (void)m_col;
    // key-token region/coord data for this lane's column (recomputed per t, key token 16*t + half)
    float inv[8];
    #pragma unroll
    for (int j = 0; j < 8; j++) {
        int n  = 16 * wv + hi * 8 + j;          // query token (padded)
        int nn = (n < NTOK) ? n : 0;
        int qr = nn / WS, qc = nn % WS;
        int ghr = wi * WS + qr, ghc = wj * WS + qc;
        int idn = (ghr < HH - WS ? 0 : (ghr < HH - SHIFT ? 1 : 2)) * 3
                + (ghc < WW - WS ? 0 : (ghc < WW - SHIFT ? 1 : 2));
        float sv[4];
        float mx = -1e30f;
        #pragma unroll
        for (int t = 0; t < 4; t++) {
            int m = 16 * t + half;              // key token
            float s;
            if (m < NTOK) {
                int kr = m / WS, kc = m % WS;
                int ridx = (qr - kr + WS - 1) * (2 * WS - 1) + (qc - kc + WS - 1);
                float bv = RPB[ridx];
                int gkr = wi * WS + kr, gkc = wj * WS + kc;
                int idm = (gkr < HH - WS ? 0 : (gkr < HH - SHIFT ? 1 : 2)) * 3
                        + (gkc < WW - WS ? 0 : (gkc < WW - SHIFT ? 1 : 2));
                s = S[t][j] * SCALE + bv + ((idn == idm) ? 0.f : -100.f);
            } else {
                s = -1e30f;
            }
            sv[t] = s;
            mx = fmaxf(mx, s);
        }
        #pragma unroll
        for (int o = 8; o >= 1; o >>= 1) mx = fmaxf(mx, __shfl_xor(mx, o, 32));
        float sum = 0.f;
        #pragma unroll
        for (int t = 0; t < 4; t++) {
            float e = __expf(sv[t] - mx);
            sum += e;
            P[n][16 * t + half] = (_Float16)e;
        }
        #pragma unroll
        for (int o = 8; o >= 1; o >>= 1) sum += __shfl_xor(sum, o, 32);
        inv[j] = 1.f / sum;
    }
    __syncthreads();

    // O = P @ V   (K-dim = 64 tokens, two 32-steps)
    v8f O[2] = {};
    #pragma unroll
    for (int ko = 0; ko < NPAD; ko += 32) {
        const _Float16* arow = &P[16 * wv + half][ko + (hi ? 8 : 0)];
        v16h a = cat16(*(const v8h*)arow, *(const v8h*)(arow + 16));
        #pragma unroll
        for (int t2 = 0; t2 < 2; t2++) {
            const _Float16* brow = &VT[16 * t2 + half][ko + (hi ? 8 : 0)];
            v16h bfrag = cat16(*(const v8h*)brow, *(const v8h*)(brow + 16));
            O[t2] = __builtin_amdgcn_wmma_f32_16x16x32_f16(
                false, a, false, bfrag, (short)0, O[t2], false, false);
        }
    }

    #pragma unroll
    for (int t2 = 0; t2 < 2; t2++) {
        #pragma unroll
        for (int j = 0; j < 8; j++) {
            int n   = 16 * wv + hi * 8 + j;
            int col = h * HD + 16 * t2 + half;
            h2[((size_t)w * NPAD + n) * CC + col] = (_Float16)(O[t2][j] * inv[j]);
        }
    }
}

// ---------------------------------------------------------------------------
// window reverse + reverse shift + residual add + LN2 (f16 out for fc1)
// grid: XROWS blocks, 192 threads
// ---------------------------------------------------------------------------
__global__ void k_reverse_ln2(const float* __restrict__ x,
                              const float* __restrict__ p,
                              const float* __restrict__ g,
                              const float* __restrict__ b,
                              float* __restrict__ x1,
                              _Float16* __restrict__ ln2h) {
    int bid = blockIdx.x;
    int c   = threadIdx.x;
    int bimg = bid / (HH * WW);
    int hw   = bid % (HH * WW);
    int oh = hw / WW, ow = hw % WW;
    int sh = (oh + HH - SHIFT) % HH, sw = (ow + WW - SHIFT) % WW;
    int wi = sh / WS, r = sh % WS, wj = sw / WS, c2 = sw % WS;
    size_t wrow = ((size_t)(bimg * NWIN + wi * 8 + wj)) * NPAD + r * WS + c2;

    float v = x[(size_t)bid * CC + c] + p[wrow * CC + c];
    x1[(size_t)bid * CC + c] = v;

    float s = v, s2 = v * v;
    #pragma unroll
    for (int o = 16; o >= 1; o >>= 1) { s += __shfl_xor(s, o, 32); s2 += __shfl_xor(s2, o, 32); }
    __shared__ float rs[6], rq[6];
    int wv = c >> 5;
    if ((c & 31) == 0) { rs[wv] = s; rq[wv] = s2; }
    __syncthreads();
    float ts = 0.f, tq = 0.f;
    #pragma unroll
    for (int i = 0; i < 6; i++) { ts += rs[i]; tq += rq[i]; }
    float mu  = ts * (1.f / CC);
    float var = tq * (1.f / CC) - mu * mu;
    float y = (v - mu) * rsqrtf(var + 1e-5f) * g[c] + b[c];
    ln2h[(size_t)bid * CC + c] = (_Float16)y;
}

// ---------------------------------------------------------------------------
extern "C" void kernel_launch(void* const* d_in, const int* in_sizes, int n_in,
                              void* d_out, int out_size, void* d_ws, size_t ws_size,
                              hipStream_t stream) {
    const float* x      = (const float*)d_in[0];
    const float* ln1_g  = (const float*)d_in[1];
    const float* ln1_b  = (const float*)d_in[2];
    const float* qkv_w  = (const float*)d_in[3];
    const float* qkv_b  = (const float*)d_in[4];
    const float* proj_w = (const float*)d_in[5];
    const float* proj_b = (const float*)d_in[6];
    const float* rpb    = (const float*)d_in[7];
    const float* ln2_g  = (const float*)d_in[8];
    const float* ln2_b  = (const float*)d_in[9];
    const float* fc1_w  = (const float*)d_in[10];
    const float* fc1_b  = (const float*)d_in[11];
    const float* fc2_w  = (const float*)d_in[12];
    const float* fc2_b  = (const float*)d_in[13];

    char* ws = (char*)d_ws;
    size_t off = 0;
    auto alloc = [&](size_t bytes) { void* p = ws + off; off += (bytes + 255) & ~(size_t)255; return p; };

    _Float16* hwin   = (_Float16*)alloc((size_t)MROWS * CC * 2);        // LN1'd windows
    _Float16* qkvbuf = (_Float16*)alloc((size_t)MROWS * 3 * CC * 2);    // qkv
    _Float16* h2     = (_Float16*)alloc((size_t)MROWS * CC * 2);        // attn out
    float*    pbuf   = (float*)   alloc((size_t)MROWS * CC * 4);        // proj out
    float*    x1     = (float*)   alloc((size_t)XROWS * CC * 4);        // residual 1
    _Float16* ln2h   = (_Float16*)alloc((size_t)XROWS * CC * 2);        // LN2 out
    _Float16* f1buf  = (_Float16*)alloc((size_t)XROWS * 4 * CC * 2);    // fc1+gelu out
    _Float16* qkvw_h = (_Float16*)alloc((size_t)3 * CC * CC * 2);
    _Float16* projw_h= (_Float16*)alloc((size_t)CC * CC * 2);
    _Float16* fc1w_h = (_Float16*)alloc((size_t)4 * CC * CC * 2);
    _Float16* fc2w_h = (_Float16*)alloc((size_t)CC * 4 * CC * 2);

    // weight converts (tiny; weights then live hot in L2)
    k_f32_to_f16<<<(3 * CC * CC + 255) / 256, 256, 0, stream>>>(qkv_w,  qkvw_h, 3 * CC * CC);
    k_f32_to_f16<<<(CC * CC + 255) / 256,     256, 0, stream>>>(proj_w, projw_h, CC * CC);
    k_f32_to_f16<<<(4 * CC * CC + 255) / 256, 256, 0, stream>>>(fc1_w,  fc1w_h, 4 * CC * CC);
    k_f32_to_f16<<<(4 * CC * CC + 255) / 256, 256, 0, stream>>>(fc2_w,  fc2w_h, 4 * CC * CC);

    // LN1 + shift + window partition
    k_ln1_partition<<<BNW * NPAD, CC, 0, stream>>>(x, ln1_g, ln1_b, hwin);

    // qkv GEMM: [131072,192] x [576,192]^T
    k_wmma_gemm<0><<<dim3(3 * CC / 64, MROWS / 128), 256, 0, stream>>>(
        hwin, qkvw_h, qkv_b, qkvbuf, nullptr, MROWS, 3 * CC, CC);

    // windowed attention
    k_window_attention<<<dim3(BNW, NH), 128, 0, stream>>>(qkvbuf, rpb, h2);

    // proj GEMM (f32 out)
    k_wmma_gemm<3><<<dim3(CC / 64, MROWS / 128), 256, 0, stream>>>(
        h2, projw_h, proj_b, pbuf, nullptr, MROWS, CC, CC);

    // window reverse + residual + LN2
    k_reverse_ln2<<<XROWS, CC, 0, stream>>>(x, pbuf, ln2_g, ln2_b, x1, ln2h);

    // fc1 GEMM + GELU
    k_wmma_gemm<1><<<dim3(4 * CC / 64, XROWS / 128), 256, 0, stream>>>(
        ln2h, fc1w_h, fc1_b, f1buf, nullptr, XROWS, 4 * CC, CC);

    // fc2 GEMM + residual -> d_out (f32)
    k_wmma_gemm<2><<<dim3(CC / 64, XROWS / 128), 256, 0, stream>>>(
        f1buf, fc2w_h, fc2_b, d_out, x1, XROWS, CC, 4 * CC);
}
